// VectorQuantizerEMA_77781857731260
// MI455X (gfx1250) — matrix operand order repaired and checked by reading, not compiled, not run on hardware
//
#include <hip/hip_runtime.h>
#include <hip/hip_bf16.h>
#include <math.h>

// ---------------------------------------------------------------------------
// VQ-VAE EMA quantizer for MI455X (gfx1250, wave32).
// Distance GEMM (65536x1024x64) via v_wmma_f32_16x16x32_bf16 (bf16 in, f32
// acc): GEMM ~1.7us vs the ~13us HBM floor from the 268MB one-hot write.
// Codebook (128KB bf16) + e_sq (4KB) staged in LDS (320KB WGP LDS); A
// fragments live in registers for all 64 column tiles. z packing is done
// through an LDS transpose so all global reads/writes are coalesced.
// ---------------------------------------------------------------------------

typedef __attribute__((ext_vector_type(16))) __bf16 v16bf;
typedef __attribute__((ext_vector_type(8)))  float  v8f;

#define NPTS   65536      // 64*32*32 points
#define KCB    1024       // codebook size
#define DDIM   64         // embedding dim
#define DECAYF 0.99f
#define EPSV   1e-10f
#define COMMITF 0.25f

// ---- pack z_e [B=64,C=64,H=32,W=32] f32 -> bf16 A-fragment layout --------
// One block per (b, hh) strip: 32 points x 64 dims. Global reads are 64
// contiguous 128B rows (coalesced); LDS transpose; the packed output for the
// two 16-row tiles of this strip is one contiguous 4KB range (coalesced).
// Layout: z_pack[((rt*2+h)*32 + lane)*16 + j] = z_flat[16*rt + (lane&15)][d],
// d = 32*h + ((lane>=16)?8:0) + (j<8 ? j : j+8)   (ISA 16-bit A 16x32)
__global__ __launch_bounds__(256) void vq_pack_z(const float* __restrict__ z_e,
                                                 __bf16* __restrict__ z_pack) {
  __shared__ float zt[2048];                         // [d=64][ww=32]
  const int bid = blockIdx.x;                        // 0..2047 = b*32 + hh
  const int t = threadIdx.x;
  const float* src = z_e + ((size_t)(bid >> 5) * 64) * 1024 + (bid & 31) * 32;
  #pragma unroll
  for (int kk = 0; kk < 8; ++kk) {                   // coalesced 128B rows
    int i = t + kk * 256;
    zt[i] = src[(size_t)(i >> 5) * 1024 + (i & 31)];
  }
  __syncthreads();

  const int o0 = t * 8;                              // 8 consecutive outputs
  const int q  = (o0 >> 10) & 1;                     // tile half within strip
  const int h  = (o0 >> 9) & 1;                      // d half
  const int l  = (o0 >> 4) & 31;                     // lane
  const int jb = o0 & 15;                            // 0 or 8
  const int wwl   = q * 16 + (l & 15);
  const int dbase = 32 * h + ((l >> 4) << 3);
  __attribute__((aligned(16))) __bf16 vals[8];
  #pragma unroll
  for (int u = 0; u < 8; ++u) {
    int j = jb + u;
    int d = dbase + (j < 8 ? j : j + 8);
    vals[u] = (__bf16)zt[d * 32 + wwl];
  }
  *(uint4*)(z_pack + (size_t)bid * 2048 + o0) = *(const uint4*)vals;
}

// ---- pack embedding [K,D] f32 -> bf16 B-fragment layout ------------------
// e_pack[((t*2+h)*32 + lane)*16 + e] = emb[16*t + (lane&15)][d],
// d = 32*h + ((lane>=16)?8:0) + (e<8 ? e : e+8)   (same k order as A)
__global__ __launch_bounds__(256) void vq_pack_e(const float* __restrict__ emb,
                                                 __bf16* __restrict__ e_pack) {
  int i = blockIdx.x * 256 + threadIdx.x;            // 0 .. K*D-1 = 65535
  int e = i & 15;
  int l = (i >> 4) & 31;
  int h = (i >> 9) & 1;
  int t = i >> 10;
  int code = t * 16 + (l & 15);
  int d    = 32 * h + ((l >> 4) << 3) + (e < 8 ? e : e + 8);
  e_pack[i] = (__bf16)emb[code * 64 + d];
}

// ---- e_sq[k] = ||emb[k]||^2 ----------------------------------------------
__global__ __launch_bounds__(256) void vq_esq(const float* __restrict__ emb,
                                              float* __restrict__ e_sq) {
  int k = blockIdx.x * 256 + threadIdx.x;            // 0..1023
  if (k < KCB) {
    float s = 0.f;
    #pragma unroll
    for (int d = 0; d < DDIM; ++d) { float v = emb[k * 64 + d]; s += v * v; }
    e_sq[k] = s;
  }
}

// ---- main: WMMA distance GEMM + argmin -----------------------------------
__global__ __launch_bounds__(256) void vq_argmin(const __bf16* __restrict__ z_pack,
                                                 const __bf16* __restrict__ e_pack,
                                                 const float* __restrict__ e_sq,
                                                 int* __restrict__ idx_out) {
  extern __shared__ char lds_raw[];                  // 128KB codebook + 4KB e_sq
  __bf16* lds_e = (__bf16*)lds_raw;
  float*  esq_l = (float*)(lds_raw + 131072);
  const int tid = threadIdx.x;

  {                                                  // cooperative LDS stage
    const uint4* src = (const uint4*)e_pack;
    uint4* dst = (uint4*)lds_raw;
    #pragma unroll
    for (int i = 0; i < 32; ++i) dst[tid + i * 256] = src[tid + i * 256];
    #pragma unroll
    for (int u = 0; u < 4; ++u) esq_l[tid + u * 256] = e_sq[tid + u * 256];
  }
  __syncthreads();

  const int wave = tid >> 5;
  const int lane = tid & 31;
  const int rt   = blockIdx.x * 8 + wave;            // row-tile 0..4095

  // A fragments for this 16-row tile (D = 0..31 and 32..63), kept in VGPRs
  v16bf a0 = *(const v16bf*)(z_pack + (((size_t)rt * 2 + 0) * 32 + lane) * 16);
  v16bf a1 = *(const v16bf*)(z_pack + (((size_t)rt * 2 + 1) * 32 + lane) * 16);

  float best[8];
  int   bidx[8];
  #pragma unroll
  for (int r = 0; r < 8; ++r) { best[r] = 3.4e38f; bidx[r] = 0x7fffffff; }

  const int col = lane & 15;
  #pragma unroll 2
  for (int t = 0; t < 64; ++t) {                     // 64 tiles of 16 codes
    v16bf b0 = *(const v16bf*)(lds_e + ((t * 2 + 0) * 32 + lane) * 16);
    v16bf b1 = *(const v16bf*)(lds_e + ((t * 2 + 1) * 32 + lane) * 16);
    int code = t * 16 + col;
    float esq = esq_l[code];
    v8f c = {};
    c = __builtin_amdgcn_wmma_f32_16x16x32_bf16(false, a0, false, b0,
                                                (short)0, c, false, false);
    c = __builtin_amdgcn_wmma_f32_16x16x32_bf16(false, a1, false, b1,
                                                (short)0, c, false, false);
    #pragma unroll
    for (int r = 0; r < 8; ++r) {                    // dist = ||e||^2 - 2 z.e
      float dist = esq - 2.0f * c[r];
      if (dist < best[r]) { best[r] = dist; bidx[r] = code; }
    }
  }

  // reduce (min, argmin) across the 16 lanes holding one row (wave32 xor)
  #pragma unroll
  for (int m = 1; m < 16; m <<= 1) {
    #pragma unroll
    for (int r = 0; r < 8; ++r) {
      float ob = __shfl_xor(best[r], m, 32);
      int   oi = __shfl_xor(bidx[r], m, 32);
      if (ob < best[r] || (ob == best[r] && oi < bidx[r])) {
        best[r] = ob; bidx[r] = oi;
      }
    }
  }
  if ((lane & 15) == 0) {                            // lane0 -> rows 0..7, lane16 -> 8..15
    int mbase = rt * 16 + ((lane >> 4) << 3);
    #pragma unroll
    for (int r = 0; r < 8; ++r) idx_out[mbase + r] = bidx[r];
  }
}

// ---- one-hot scatter + counts --------------------------------------------
__global__ __launch_bounds__(256) void vq_onehot(const int* __restrict__ idx,
                                                 float* __restrict__ enc,
                                                 float* __restrict__ counts) {
  int n = blockIdx.x * 256 + threadIdx.x;            // 0..65535
  int k = idx[n];
  enc[(size_t)n * KCB + k] = 1.0f;
  atomicAdd(&counts[k], 1.0f);
}

// ---- per-element: quantize, straight-through, loss partial, dw atomics ---
__global__ __launch_bounds__(256) void vq_elem(const float* __restrict__ z_e,
                                               const float* __restrict__ emb,
                                               const int* __restrict__ idx,
                                               float* __restrict__ q_out,
                                               float* __restrict__ dw,
                                               float* __restrict__ loss_acc) {
  __shared__ float red[256];
  int i = blockIdx.x * 256 + threadIdx.x;            // 0..4194303 (BCHW linear)
  int ww = i & 31;
  int hh = (i >> 5) & 31;
  int d  = (i >> 10) & 63;
  int b  = i >> 16;
  int n  = (b << 10) + (hh << 5) + ww;
  int k  = idx[n];
  float q = emb[k * 64 + d];
  float z = z_e[i];
  float diff = q - z;
  q_out[i] = z + diff;                               // straight-through fwd
  atomicAdd(&dw[k * 64 + d], z);                     // dw = enc^T @ z_flat
  red[threadIdx.x] = diff * diff;
  __syncthreads();
  for (int s = 128; s > 0; s >>= 1) {
    if (threadIdx.x < s) red[threadIdx.x] += red[threadIdx.x + s];
    __syncthreads();
  }
  if (threadIdx.x == 0) atomicAdd(loss_acc, red[0]);
}

// ---- EMA finalize + perplexity (single block, 1024 threads) --------------
__global__ __launch_bounds__(1024) void vq_final(const float* __restrict__ ema_cs,
                                                 const float* __restrict__ ema_w,
                                                 const float* __restrict__ counts,
                                                 const float* __restrict__ dw,
                                                 const float* __restrict__ loss_acc,
                                                 float* __restrict__ new_cs,
                                                 float* __restrict__ new_ema,
                                                 float* __restrict__ new_emb,
                                                 float* __restrict__ loss_out,
                                                 float* __restrict__ perp_out) {
  __shared__ float sh[1024];
  int k = threadIdx.x;
  float cnt = counts[k];
  float ncs = DECAYF * ema_cs[k] + (1.0f - DECAYF) * cnt;
  new_cs[k] = ncs;

  sh[k] = ncs; __syncthreads();
  for (int s = 512; s > 0; s >>= 1) {
    if (k < s) sh[k] += sh[k + s];
    __syncthreads();
  }
  float nsum = sh[0]; __syncthreads();

  float p = cnt * (1.0f / (float)NPTS);
  sh[k] = p * logf(p + 1e-10f); __syncthreads();
  for (int s = 512; s > 0; s >>= 1) {
    if (k < s) sh[k] += sh[k + s];
    __syncthreads();
  }
  if (k == 0) {
    perp_out[0] = expf(-sh[0]);
    loss_out[0] = COMMITF * loss_acc[0] / (float)(NPTS * DDIM);
  }

  float csize = (ncs + EPSV) / (nsum + (float)KCB * EPSV) * nsum;
  #pragma unroll 4
  for (int d = 0; d < DDIM; ++d) {
    float nw = DECAYF * ema_w[k * 64 + d] + (1.0f - DECAYF) * dw[k * 64 + d];
    new_ema[k * 64 + d] = nw;
    new_emb[k * 64 + d] = nw / csize;
  }
}

// ---------------------------------------------------------------------------
extern "C" void kernel_launch(void* const* d_in, const int* in_sizes, int n_in,
                              void* d_out, int out_size, void* d_ws, size_t ws_size,
                              hipStream_t stream) {
  const float* z_e    = (const float*)d_in[0];
  const float* emb    = (const float*)d_in[1];
  const float* ema_cs = (const float*)d_in[2];
  const float* ema_w  = (const float*)d_in[3];

  float* out      = (float*)d_out;
  float* q_out    = out;                         // 4,194,304
  float* loss_out = out + 4194304;               // 1
  float* perp_out = out + 4194305;               // 1
  float* enc      = out + 4194306;               // 67,108,864
  float* new_emb  = out + 71303170;              // 65,536
  float* new_cs   = out + 71368706;              // 1,024
  float* new_ema  = out + 71369730;              // 65,536

  char* ws = (char*)d_ws;
  __bf16* z_pack  = (__bf16*)(ws);               // 8 MB
  __bf16* e_pack  = (__bf16*)(ws + 8388608);     // 128 KB
  float*  e_sq    = (float*) (ws + 8519680);     // 4 KB
  int*    idx     = (int*)   (ws + 8523776);     // 256 KB
  float*  counts  = (float*) (ws + 8785920);     // 4 KB
  float*  dw      = (float*) (ws + 8790016);     // 256 KB
  float*  loss_ac = (float*) (ws + 9052160);     // 16 B

  // zero accumulators + one-hot output (harness poisons buffers)
  hipMemsetAsync(enc,     0, (size_t)67108864 * sizeof(float), stream);
  hipMemsetAsync(counts,  0, KCB * sizeof(float), stream);
  hipMemsetAsync(dw,      0, KCB * DDIM * sizeof(float), stream);
  hipMemsetAsync(loss_ac, 0, 16, stream);

  vq_pack_z<<<2048,  256, 0, stream>>>(z_e, z_pack);
  vq_pack_e<<<256,   256, 0, stream>>>(emb, e_pack);
  vq_esq  <<<4,      256, 0, stream>>>(emb, e_sq);
  vq_argmin<<<512,   256, 135168, stream>>>(z_pack, e_pack, e_sq, idx);
  vq_onehot<<<256,   256, 0, stream>>>(idx, enc, counts);
  vq_elem <<<16384,  256, 0, stream>>>(z_e, emb, idx, q_out, dw, loss_ac);
  vq_final<<<1,     1024, 0, stream>>>(ema_cs, ema_w, counts, dw, loss_ac,
                                       new_cs, new_ema, new_emb, loss_out, perp_out);
}
